// RelativeAttention_5119601017479
// MI455X (gfx1250) — matrix operand order repaired and checked by reading, not compiled, not run on hardware
//
#include <hip/hip_runtime.h>
#include <hip/hip_bf16.h>
#include <math.h>

// Problem constants (from reference): Q=KL=1024, B=4, N=16, D=64, R=2048
#define QLEN 1024
#define KLEN 1024
#define BB   4
#define NN   16
#define DD   64
#define RR   2048
#define QT   64     // q rows per workgroup
#define KT   64     // k cols per inner tile
#define NTHREADS 128

typedef __bf16 bf16;
typedef __attribute__((ext_vector_type(2)))  __bf16 v2bf;
typedef __attribute__((ext_vector_type(8)))  __bf16 v8bf;
typedef __attribute__((ext_vector_type(16))) __bf16 v16bf;
typedef __attribute__((ext_vector_type(8)))  float  v8f;

#if defined(__has_builtin)
#  if __has_builtin(__builtin_amdgcn_cvt_pk_bf16_f32)
#    define USE_PK_CVT 1
#  endif
#endif

__device__ __forceinline__ bf16 f2bf(float f) {
  union { float f; unsigned u; } x; x.f = f;
  unsigned r = x.u + 0x7FFFu + ((x.u >> 16) & 1u);   // round-to-nearest-even
  unsigned short h = (unsigned short)(r >> 16);
  union { unsigned short u; bf16 b; } y; y.u = h;
  return y.b;
}

__device__ __forceinline__ v2bf pk2(float a, float b) {
#ifdef USE_PK_CVT
  return __builtin_amdgcn_cvt_pk_bf16_f32(a, b);
#else
  v2bf r; r[0] = f2bf(a); r[1] = f2bf(b); return r;
#endif
}

// ---- WMMA fragment loaders (CDNA5 v_wmma_f32_16x16x32_bf16 layouts, ISA 7.12.2) ----

// A operand (16x32, bf16) from row-major LDS tile.
// lane<16: elems 0..7 = K[kb..kb+7],  elems 8..15 = K[kb+16..kb+23]; lane>=16: +8 K offset.
__device__ __forceinline__ v16bf ld_fragA(const bf16* base, int stride, int row0, int kb, int lane) {
  int m  = row0 + (lane & 15);
  int ko = kb + ((lane >> 4) << 3);
  v8bf lo = *(const v8bf*)(base + m * stride + ko);
  v8bf hi = *(const v8bf*)(base + m * stride + ko + 16);
  return __builtin_shufflevector(lo, hi, 0,1,2,3,4,5,6,7,8,9,10,11,12,13,14,15);
}

// B operand (32x16, bf16) from row-major *transposed* LDS tile Bt[n][k].
// lane<16: elems 0..7 = K[kb..kb+7], elems 8..15 = K[kb+16..kb+23]; lane>=16: +8 K offset.
__device__ __forceinline__ v16bf ld_fragB(const bf16* base, int stride, int col0, int kb, int lane) {
  int n  = col0 + (lane & 15);
  int ko = kb + ((lane >> 4) << 3);
  const bf16* p = base + n * stride + ko;
  v8bf lo = *(const v8bf*)(p);
  v8bf hi = *(const v8bf*)(p + 16);
  return __builtin_shufflevector(lo, hi, 0,1,2,3,4,5,6,7,8,9,10,11,12,13,14,15);
}

// C/D accumulator (16x16 f32) store: lane<16 -> M=v, lane>=16 -> M=v+8; N = lane&15.
__device__ __forceinline__ void st_acc(float* base, int stride, int row0, int col0, v8f c, int lane) {
  int rr = row0 + ((lane >> 4) << 3);
  int cc = col0 + (lane & 15);
#pragma unroll
  for (int v = 0; v < 8; ++v) base[(rr + v) * stride + cc] = c[v];
}

#define LDS  65    // f32 stride for sS: bank = (m + j) % 64 -> conflict-free row access
#define LDB  132   // f32 stride for 128-wide band: diagonal reads conflict-free
#define LDP  72    // bf16 stride for sP: 144B rows (16B aligned), mild stagger

__global__ __launch_bounds__(NTHREADS)
void xlnet_rel_attn_kernel(const float* __restrict__ q_head,
                           const float* __restrict__ k_head_h,
                           const float* __restrict__ v_head_h,
                           const float* __restrict__ k_head_r,
                           const float* __restrict__ seg_embed,
                           const float* __restrict__ r_w_bias,
                           const float* __restrict__ r_r_bias,
                           const float* __restrict__ r_s_bias,
                           const float* __restrict__ attn_mask,
                           const unsigned char* __restrict__ seg_mat,
                           float* __restrict__ out)
{
  __shared__ bf16  sQw[QT][DD];     // q + r_w_bias
  __shared__ bf16  sQr[QT][DD];     // q + r_r_bias
  __shared__ bf16  sK [KT][DD];     // k_head_h tile (rows = k index)
  __shared__ bf16  sVt[DD][KT];     // v tile, transposed: sVt[d][j]
  __shared__ bf16  sKR[128][DD];    // k_head_r band rows
  __shared__ float sS [QT][LDS];    // AC scores (f32), WMMA -> softmax handoff
  __shared__ float sBD[QT][LDB];    // BD band scores (f32)
  __shared__ bf16  sP [QT][LDP];    // probabilities (bf16) for PV WMMA
  __shared__ float sEf0[QT], sEf1[QT], sM[QT], sL[QT], sC[QT];
  __shared__ float sRmax[2][QT], sRsum[2][QT];

  const int tid  = threadIdx.x;
  const int lane = tid & 31;
  const int wave = tid >> 5;
  const int gid  = blockIdx.x;
  const int qt = gid & 15;          // q tile index
  const int n  = (gid >> 4) & 15;   // head
  const int b  = gid >> 8;          // batch
  const int i0 = qt * QT;
  const int m0 = wave * 16;         // this wave's q-row strip

  // softmax role of this thread (2 threads per q row, 32 cols each)
  const int sm_row  = tid & 63;
  const int sm_c0   = (tid >> 6) * 32;
  const int sm_half = tid >> 6;

  v8f oacc[4] = {};                 // O accumulator strip [16 x 64] in registers

  // ---------------- stage Q (+biases), EF term, init stats ----------------
  if (tid < QT) {
    const int m = tid;
    const int i = i0 + m;
    const float* qrow = q_head + (((size_t)i * BB + b) * NN + n) * DD;
    const float* wb  = r_w_bias + n * DD;
    const float* rb  = r_r_bias + n * DD;
    const float* sb  = r_s_bias + n * DD;
    const float* se0 = seg_embed + (0 * NN + n) * DD;
    const float* se1 = seg_embed + (1 * NN + n) * DD;
    float e0 = 0.f, e1 = 0.f;
#pragma unroll 4
    for (int d = 0; d < DD; d += 2) {
      float q0 = qrow[d], q1 = qrow[d + 1];
      *(v2bf*)&sQw[m][d] = pk2(q0 + wb[d], q1 + wb[d + 1]);
      *(v2bf*)&sQr[m][d] = pk2(q0 + rb[d], q1 + rb[d + 1]);
      float s0 = q0 + sb[d], s1 = q1 + sb[d + 1];
      e0 += s0 * se0[d] + s1 * se0[d + 1];
      e1 += s0 * se1[d] + s1 * se1[d + 1];
    }
    sEf0[m] = e0; sEf1[m] = e1;
    sM[m] = -INFINITY; sL[m] = 0.f;
  }
  __syncthreads();

  // Q fragments are loop-invariant: hoist to registers once (saves 8 b128 DS loads/iter).
  const v16bf aw0 = ld_fragA(&sQw[0][0], DD, m0, 0,  lane);
  const v16bf aw1 = ld_fragA(&sQw[0][0], DD, m0, 32, lane);
  const v16bf ar0 = ld_fragA(&sQr[0][0], DD, m0, 0,  lane);
  const v16bf ar1 = ld_fragA(&sQr[0][0], DD, m0, 32, lane);

  // ---------------- k-tile loop (flash-attention style) ----------------
  for (int j0 = 0; j0 < KLEN; j0 += KT) {
    // ----- preload this tile's mask/seg gathers into registers: their (strided,
    //       16B-pitch) latency then overlaps the whole WMMA score phase -----
    float mreg[32];
    unsigned char sgreg[32];
    {
      const float* mrow = attn_mask +
          ((size_t)(i0 + sm_row) * KLEN + j0 + sm_c0) * BB + b;
      const unsigned char* srow = seg_mat +
          ((size_t)(i0 + sm_row) * KLEN + j0 + sm_c0) * BB + b;
#pragma unroll
      for (int j = 0; j < 32; ++j) {
        mreg[j]  = mrow[(size_t)j * BB];   // RT loads: reused by all 16 heads via L2
        sgreg[j] = srow[(size_t)j * BB];
      }
    }

    // ----- stage K, V(transposed), KR band; prefetch next tile -----
    {
      const int r  = tid >> 1;
      const int c0 = (tid & 1) * 32;
      const float* krow = k_head_h + (((size_t)(j0 + r) * BB + b) * NN + n) * DD + c0;
      const float* vrow = v_head_h + (((size_t)(j0 + r) * BB + b) * NN + n) * DD + c0;
#pragma unroll 4
      for (int d = 0; d < 32; d += 2) {
        *(v2bf*)&sK[r][c0 + d] = pk2(krow[d], krow[d + 1]);
        v2bf vv = pk2(vrow[d], vrow[d + 1]);
        sVt[c0 + d][r]     = vv[0];
        sVt[c0 + d + 1][r] = vv[1];
      }
      // stage k_head_r band: one row per thread. rel-shift: row = j + KL - i
      int kr  = j0 + (KLEN - QT + 1) - i0 + tid;     // base = j0 + 961 - i0
      int krc = kr < RR ? kr : RR - 1;               // last band col never consumed
      const float* rrow = k_head_r + (((size_t)krc * BB + b) * NN + n) * DD;
#pragma unroll 4
      for (int d = 0; d < DD; d += 2)
        *(v2bf*)&sKR[tid][d] = pk2(rrow[d], rrow[d + 1]);

      if (j0 + KT < KLEN) {
        __builtin_prefetch(krow + (size_t)KT * BB * NN * DD, 0, 3);
        __builtin_prefetch(vrow + (size_t)KT * BB * NN * DD, 0, 3);
        int krn = krc + KT < RR ? krc + KT : RR - 1;
        __builtin_prefetch(k_head_r + (((size_t)krn * BB + b) * NN + n) * DD, 0, 3);
      }
    }
    __syncthreads();

    // ----- WMMA: AC tile [64x64] and BD band [64x128]; wave owns 16-row strip -----
    {
#pragma unroll
      for (int nt = 0; nt < 4; ++nt) {
        v16bf b0 = ld_fragB(&sK[0][0], DD, nt * 16, 0,  lane);
        v16bf b1 = ld_fragB(&sK[0][0], DD, nt * 16, 32, lane);
        v8f c = {};
        c = __builtin_amdgcn_wmma_f32_16x16x32_bf16(false, aw0, false, b0, (short)0, c, false, false);
        c = __builtin_amdgcn_wmma_f32_16x16x32_bf16(false, aw1, false, b1, (short)0, c, false, false);
        st_acc(&sS[0][0], LDS, m0, nt * 16, c, lane);
      }
#pragma unroll
      for (int tt = 0; tt < 8; ++tt) {
        v16bf b0 = ld_fragB(&sKR[0][0], DD, tt * 16, 0,  lane);
        v16bf b1 = ld_fragB(&sKR[0][0], DD, tt * 16, 32, lane);
        v8f c = {};
        c = __builtin_amdgcn_wmma_f32_16x16x32_bf16(false, ar0, false, b0, (short)0, c, false, false);
        c = __builtin_amdgcn_wmma_f32_16x16x32_bf16(false, ar1, false, b1, (short)0, c, false, false);
        st_acc(&sBD[0][0], LDB, m0, tt * 16, c, lane);
      }
    }
    __syncthreads();

    // ----- combine scores + online softmax: 2 threads per row, 32 cols each -----
    {
      const int m  = sm_row;
      const int c0 = sm_c0;
      const float e0 = sEf0[m], e1 = sEf1[m];
      float sreg[32];                      // combined scores live in registers
      float mx = -INFINITY;
#pragma unroll
      for (int j = 0; j < 32; ++j) {
        int jj = c0 + j;
        float sc = sS[m][jj] + sBD[m][jj - m + (QT - 1)] + (sgreg[j] ? e1 : e0);
        sc = sc * 0.125f - 1e30f * mreg[j];
        sreg[j] = sc;
        mx = fmaxf(mx, sc);
      }
      sRmax[sm_half][m] = mx;
      __syncthreads();
      float newm = fmaxf(sM[m], fmaxf(sRmax[0][m], sRmax[1][m]));
      float sum = 0.f;
#pragma unroll
      for (int j = 0; j < 32; j += 2) {
        float p0 = __expf(sreg[j]     - newm);
        float p1 = __expf(sreg[j + 1] - newm);
        *(v2bf*)&sP[m][c0 + j] = pk2(p0, p1);
        sum += p0 + p1;
      }
      sRsum[sm_half][m] = sum;
      __syncthreads();
      if (sm_half == 0) {
        float corr = __expf(sM[m] - newm);
        sL[m] = sL[m] * corr + sRsum[0][m] + sRsum[1][m];
        sM[m] = newm;
        sC[m] = corr;
      }
    }
    __syncthreads();

    // ----- rescale O registers by per-row correction, then O += P x V -----
    {
      const int rbase = m0 + ((lane >> 4) << 3);   // C-layout: lane half selects +8 rows
      float cr[8];
#pragma unroll
      for (int v = 0; v < 8; ++v) cr[v] = sC[rbase + v];
#pragma unroll
      for (int dt = 0; dt < 4; ++dt)
#pragma unroll
        for (int v = 0; v < 8; ++v) oacc[dt][v] *= cr[v];

      v16bf a0 = ld_fragA(&sP[0][0], LDP, m0, 0,  lane);
      v16bf a1 = ld_fragA(&sP[0][0], LDP, m0, 32, lane);
#pragma unroll
      for (int dt = 0; dt < 4; ++dt) {
        v16bf b0 = ld_fragB(&sVt[0][0], KT, dt * 16, 0,  lane);
        v16bf b1 = ld_fragB(&sVt[0][0], KT, dt * 16, 32, lane);
        oacc[dt] = __builtin_amdgcn_wmma_f32_16x16x32_bf16(false, a0, false, b0, (short)0, oacc[dt], false, false);
        oacc[dt] = __builtin_amdgcn_wmma_f32_16x16x32_bf16(false, a1, false, b1, (short)0, oacc[dt], false, false);
      }
    }
    __syncthreads();
  }

  // ---------------- epilogue: divide by l and store straight from registers ----------------
  {
    const int rbase = m0 + ((lane >> 4) << 3);
    const int col   = lane & 15;
    float inv[8];
#pragma unroll
    for (int v = 0; v < 8; ++v) inv[v] = 1.0f / sL[rbase + v];
#pragma unroll
    for (int dt = 0; dt < 4; ++dt) {
#pragma unroll
      for (int v = 0; v < 8; ++v) {
        int i = i0 + rbase + v;
        out[(((size_t)i * BB + b) * NN + n) * DD + dt * 16 + col] = oacc[dt][v] * inv[v];
      }
    }
  }
}

extern "C" void kernel_launch(void* const* d_in, const int* in_sizes, int n_in,
                              void* d_out, int out_size, void* d_ws, size_t ws_size,
                              hipStream_t stream) {
  (void)in_sizes; (void)n_in; (void)out_size; (void)d_ws; (void)ws_size;
  const float* q_head    = (const float*)d_in[0];
  const float* k_head_h  = (const float*)d_in[1];
  const float* v_head_h  = (const float*)d_in[2];
  const float* k_head_r  = (const float*)d_in[3];
  const float* seg_embed = (const float*)d_in[4];
  const float* r_w_bias  = (const float*)d_in[5];
  const float* r_r_bias  = (const float*)d_in[6];
  const float* r_s_bias  = (const float*)d_in[7];
  const float* attn_mask = (const float*)d_in[8];
  const unsigned char* seg_mat = (const unsigned char*)d_in[9];  // jnp.bool_ = 1 byte
  float* out = (float*)d_out;

  dim3 grid(BB * NN * (QLEN / QT));   // 4*16*16 = 1024 workgroups: one per (b, n, q-tile)
  xlnet_rel_attn_kernel<<<grid, NTHREADS, 0, stream>>>(
      q_head, k_head_h, v_head_h, k_head_r, seg_embed,
      r_w_bias, r_r_bias, r_s_bias, attn_mask, seg_mat, out);
}
